// VectorizedSlidingWindowAttention_21268678049965
// MI455X (gfx1250) — compile-verified, matrix operand order checked by reading, not run on hardware
//
#include <hip/hip_runtime.h>
#include <hip/hip_bf16.h>
#include <math.h>

typedef __attribute__((ext_vector_type(16))) __bf16 v16bf;
typedef __attribute__((ext_vector_type(8)))  float  v8f;
typedef __attribute__((ext_vector_type(8)))  __bf16 v8bf;

#define DIM   256
#define NH    8
#define HD    32
#define NTOK  64
#define GH    31
#define GW    31
#define BATCH 2
#define HIMG  128
#define WIMG  128
#define QKVN  (3 * DIM)   // 768

// ---------------------------------------------------------------------------
// Kernel 0: one-shot weight conversion f32 -> bf16 (keeps GEMM loops pure
// bf16 16-byte loads; weights then live in L2 as bf16).
// ---------------------------------------------------------------------------
__global__ __launch_bounds__(256)
void cvt_weights(const float* __restrict__ qw, const float* __restrict__ pw,
                 __bf16* __restrict__ qwb, __bf16* __restrict__ pwb)
{
    int i = blockIdx.x * 256 + threadIdx.x;
    if (i < QKVN * DIM) qwb[i] = (__bf16)qw[i];
    if (i < DIM * DIM)  pwb[i] = (__bf16)pw[i];
}

// ---------------------------------------------------------------------------
// Kernel 1: qkv[M,768](bf16) = x[M,256](f32) @ qkv_w[768,256]^T + qkv_b
// One block per 16-row M-slab; A staged once in LDS as bf16 (8KB);
// 8 waves x 6 N-tiles cover all 768 columns -> x is read exactly once.
// ---------------------------------------------------------------------------
__global__ __launch_bounds__(256)
void qkv_gemm(const float* __restrict__ x, const __bf16* __restrict__ wb,
              const float* __restrict__ bias, __bf16* __restrict__ qkv)
{
    __shared__ __bf16 alds[16 * DIM];          // 8 KB staged A slab
    const int tid = threadIdx.x;
    const int m0  = blockIdx.x * 16;

    const float* src = x + (size_t)m0 * DIM;
#pragma unroll
    for (int i = 0; i < 16; ++i)
        alds[i * DIM + tid] = (__bf16)src[i * DIM + tid];
    __syncthreads();

    const int lane = tid & 31;
    const int wave = tid >> 5;
    const int l16  = lane & 15;
    const int half = lane >> 4;
    const int kb   = half * 8;
    const __bf16* arow = alds + l16 * DIM;

    v8f acc[6] = {};
    for (int k = 0; k < DIM; k += 32) {
        v16bf a;
        v8bf alo = *(const v8bf*)(arow + k + kb);
        v8bf ahi = *(const v8bf*)(arow + k + kb + 16);
#pragma unroll
        for (int i = 0; i < 8; ++i) { a[i] = alo[i]; a[i + 8] = ahi[i]; }
#pragma unroll
        for (int nt = 0; nt < 6; ++nt) {
            const __bf16* brow = wb + (size_t)(wave * 96 + nt * 16 + l16) * DIM;
            v16bf b;
            v8bf blo = *(const v8bf*)(brow + k + kb);
            v8bf bhi = *(const v8bf*)(brow + k + kb + 16);
#pragma unroll
            for (int i = 0; i < 8; ++i) { b[i] = blo[i]; b[i + 8] = bhi[i]; }
            acc[nt] = __builtin_amdgcn_wmma_f32_16x16x32_bf16(false, a, false, b,
                                                              (short)0, acc[nt], false, false);
        }
    }
#pragma unroll
    for (int nt = 0; nt < 6; ++nt) {
        int n = wave * 96 + nt * 16 + l16;
        float bv = bias[n];
#pragma unroll
        for (int r = 0; r < 8; ++r)
            qkv[(size_t)(m0 + r + 8 * half) * QKVN + n] = (__bf16)(acc[nt][r] + bv);
    }
}

// ---------------------------------------------------------------------------
// Kernel 2: one wave per (window, head). S = rope(Q) rope(K)^T via WMMA,
// softmax in LDS, O = P V via WMMA, atomic scatter-add into merge buffer.
// ---------------------------------------------------------------------------
__global__ __launch_bounds__(32)
void win_attn(const __bf16* __restrict__ qkv, float* __restrict__ accimg)
{
    __shared__ float smem[NTOK * NTOK];        // 16 KB: S (f32) then P (bf16)
    __bf16* pmem = (__bf16*)smem;

    const int task = blockIdx.x;
    const int head = task & 7;
    const int widx = task >> 3;
    const int gx = widx % GW;
    const int gy = (widx / GW) % GH;
    const int b  = widx / (GW * GH);
    const int h0 = gy * 4, w0 = gx * 4;

    const int lane = threadIdx.x;
    const int l16  = lane & 15;
    const int half = lane >> 4;
    const int kb   = half * 8;

    auto rowof = [&](int t) -> size_t {
        int hh = h0 + (t >> 3), ww = w0 + (t & 7);
        return (size_t)((b * HIMG + hh) * WIMG + ww);
    };

    const int qcol = head * HD;
    const int kcol = DIM + head * HD;
    const int vcol = 2 * DIM + head * HD;
    const float LOGF = 0.57564627325f;   // ln(10000)/16

    // ---- K fragments (B-matrix, 4 N-tiles) with in-lane RoPE ----
    v16bf ktile[4];
#pragma unroll
    for (int ni = 0; ni < 4; ++ni) {
        int t = ni * 16 + l16;                     // key token position
        const __bf16* srcp = qkv + rowof(t) * QKVN + kcol;
        float fn = (float)t;
#pragma unroll
        for (int i = 0; i < 8; ++i) {
            int d1 = kb + i;                       // rotation pair (d1, d1+16)
            float f1 = (float)srcp[d1];
            float f2 = (float)srcp[d1 + 16];
            float th = fn * __expf(-LOGF * (float)d1);
            float s, c; __sincosf(th, &s, &c);
            ktile[ni][i]     = (__bf16)(f1 * c - f2 * s);
            ktile[ni][i + 8] = (__bf16)(f2 * c + f1 * s);
        }
    }

    // ---- S = Q K^T, tile by tile, into LDS ----
#pragma unroll
    for (int mi = 0; mi < 4; ++mi) {
        int t = mi * 16 + l16;                     // query token position
        const __bf16* srcp = qkv + rowof(t) * QKVN + qcol;
        float fn = (float)t;
        v16bf qa;
#pragma unroll
        for (int i = 0; i < 8; ++i) {
            int d1 = kb + i;
            float f1 = (float)srcp[d1];
            float f2 = (float)srcp[d1 + 16];
            float th = fn * __expf(-LOGF * (float)d1);
            float s, c; __sincosf(th, &s, &c);
            qa[i]     = (__bf16)(f1 * c - f2 * s);
            qa[i + 8] = (__bf16)(f2 * c + f1 * s);
        }
#pragma unroll
        for (int ni = 0; ni < 4; ++ni) {
            v8f sacc = {};
            sacc = __builtin_amdgcn_wmma_f32_16x16x32_bf16(false, qa, false, ktile[ni],
                                                           (short)0, sacc, false, false);
#pragma unroll
            for (int r = 0; r < 8; ++r)
                smem[(mi * 16 + r + 8 * half) * NTOK + ni * 16 + l16] = sacc[r];
        }
    }
    __syncthreads();

    // ---- row softmax; write P (bf16) over the already-consumed prefix ----
    const float scale = 0.17677669529f;            // 1/sqrt(32)
    for (int row = 0; row < NTOK; ++row) {
        float s0 = smem[row * NTOK + lane] * scale;
        float s1 = smem[row * NTOK + lane + 32] * scale;
        float mx = fmaxf(s0, s1);
#pragma unroll
        for (int off = 16; off > 0; off >>= 1)
            mx = fmaxf(mx, __shfl_xor(mx, off, 32));
        float e0 = __expf(s0 - mx);
        float e1 = __expf(s1 - mx);
        float sm = e0 + e1;
#pragma unroll
        for (int off = 16; off > 0; off >>= 1)
            sm += __shfl_xor(sm, off, 32);
        float inv = __frcp_rn(sm);
        // P row bytes [128*row,128*row+128) alias S row row/2 (already consumed)
        pmem[row * NTOK + lane]      = (__bf16)(e0 * inv);
        pmem[row * NTOK + lane + 32] = (__bf16)(e1 * inv);
    }
    __syncthreads();

    // ---- V fragments (B-matrix): 2 K-tiles x 2 N-tiles, kept in regs ----
    v16bf vb[2][2];
#pragma unroll
    for (int kt = 0; kt < 2; ++kt) {
#pragma unroll
        for (int nt = 0; nt < 2; ++nt) {
            int n = nt * 16 + l16;                 // head-dim column
#pragma unroll
            for (int i = 0; i < 8; ++i) {
                int t1 = kt * 32 + kb + i;
                vb[kt][nt][i]     = qkv[rowof(t1) * QKVN + vcol + n];
                vb[kt][nt][i + 8] = qkv[rowof(t1 + 16) * QKVN + vcol + n];
            }
        }
    }

    // ---- O = P V ; scatter-add into merge accumulator ----
#pragma unroll
    for (int mi = 0; mi < 4; ++mi) {
        v8f oacc[2] = { v8f{}, v8f{} };
        int m = mi * 16 + l16;
#pragma unroll
        for (int kt = 0; kt < 2; ++kt) {
            v16bf pa;
            v8bf lo = *(const v8bf*)&pmem[m * NTOK + kt * 32 + kb];
            v8bf hi = *(const v8bf*)&pmem[m * NTOK + kt * 32 + kb + 16];
#pragma unroll
            for (int i = 0; i < 8; ++i) { pa[i] = lo[i]; pa[i + 8] = hi[i]; }
#pragma unroll
            for (int nt = 0; nt < 2; ++nt)
                oacc[nt] = __builtin_amdgcn_wmma_f32_16x16x32_bf16(false, pa, false, vb[kt][nt],
                                                                   (short)0, oacc[nt], false, false);
        }
#pragma unroll
        for (int nt = 0; nt < 2; ++nt) {
#pragma unroll
            for (int r = 0; r < 8; ++r) {
                int t = mi * 16 + r + 8 * half;
                int c = head * HD + nt * 16 + l16;
                unsafeAtomicAdd(accimg + rowof(t) * DIM + c, oacc[nt][r]);
            }
        }
    }
}

// ---------------------------------------------------------------------------
// Kernel 3: out[M,256](f32) = (acc/cnt)[M,256] @ proj_w[256,256]^T + proj_b
// Overlap count is closed-form: c(h)=2 for 4<=h<=123 else 1; applied once at
// LDS stage time. One block per 16-row slab; 8 waves x 2 N-tiles.
// ---------------------------------------------------------------------------
__global__ __launch_bounds__(256)
void proj_gemm(const float* __restrict__ accimg, const __bf16* __restrict__ wb,
               const float* __restrict__ bias, float* __restrict__ out)
{
    __shared__ __bf16 alds[16 * DIM];          // 8 KB staged (acc/cnt) slab
    const int tid = threadIdx.x;
    const int m0  = blockIdx.x * 16;

#pragma unroll
    for (int i = 0; i < 16; ++i) {
        int pix = m0 + i;
        int hh = (pix / WIMG) % HIMG;
        int ww = pix % WIMG;
        float ch = (hh >= 4 && hh <= 123) ? 2.f : 1.f;
        float cw = (ww >= 4 && ww <= 123) ? 2.f : 1.f;
        float rinv = 1.f / (ch * cw);
        alds[i * DIM + tid] = (__bf16)(accimg[(size_t)pix * DIM + tid] * rinv);
    }
    __syncthreads();

    const int lane = tid & 31;
    const int wave = tid >> 5;
    const int l16  = lane & 15;
    const int half = lane >> 4;
    const int kb   = half * 8;
    const __bf16* arow = alds + l16 * DIM;

    v8f acc[2] = {};
    for (int k = 0; k < DIM; k += 32) {
        v16bf a;
        v8bf alo = *(const v8bf*)(arow + k + kb);
        v8bf ahi = *(const v8bf*)(arow + k + kb + 16);
#pragma unroll
        for (int i = 0; i < 8; ++i) { a[i] = alo[i]; a[i + 8] = ahi[i]; }
#pragma unroll
        for (int nt = 0; nt < 2; ++nt) {
            const __bf16* brow = wb + (size_t)(wave * 32 + nt * 16 + l16) * DIM;
            v16bf b;
            v8bf blo = *(const v8bf*)(brow + k + kb);
            v8bf bhi = *(const v8bf*)(brow + k + kb + 16);
#pragma unroll
            for (int i = 0; i < 8; ++i) { b[i] = blo[i]; b[i + 8] = bhi[i]; }
            acc[nt] = __builtin_amdgcn_wmma_f32_16x16x32_bf16(false, a, false, b,
                                                              (short)0, acc[nt], false, false);
        }
    }
#pragma unroll
    for (int nt = 0; nt < 2; ++nt) {
        int n = wave * 32 + nt * 16 + l16;
        float bv = bias[n];
#pragma unroll
        for (int r = 0; r < 8; ++r)
            out[(size_t)(m0 + r + 8 * half) * DIM + n] = acc[nt][r] + bv;
    }
}

// ---------------------------------------------------------------------------
extern "C" void kernel_launch(void* const* d_in, const int* in_sizes, int n_in,
                              void* d_out, int out_size, void* d_ws, size_t ws_size,
                              hipStream_t stream)
{
    const float* x      = (const float*)d_in[0];
    const float* qkv_w  = (const float*)d_in[1];
    const float* qkv_b  = (const float*)d_in[2];
    const float* proj_w = (const float*)d_in[3];
    const float* proj_b = (const float*)d_in[4];
    float* out = (float*)d_out;

    const int M = BATCH * HIMG * WIMG;             // 32768 pixels
    char* ws = (char*)d_ws;
    float*  accimg = (float*)ws;                         // 33.5 MB
    ws += (size_t)M * DIM * sizeof(float);
    __bf16* qkv = (__bf16*)ws;                           // 50.3 MB
    ws += (size_t)M * QKVN * sizeof(__bf16);
    __bf16* qwb = (__bf16*)ws;                           // 0.39 MB
    ws += (size_t)QKVN * DIM * sizeof(__bf16);
    __bf16* pwb = (__bf16*)ws;                           // 0.13 MB

    hipMemsetAsync(accimg, 0, (size_t)M * DIM * sizeof(float), stream);

    cvt_weights<<<dim3((QKVN * DIM + 255) / 256), dim3(256), 0, stream>>>(qkv_w, proj_w, qwb, pwb);

    qkv_gemm<<<dim3(M / 16), dim3(256), 0, stream>>>(x, qwb, qkv_b, qkv);

    win_attn<<<dim3(BATCH * GH * GW * NH), dim3(32), 0, stream>>>(qkv, accimg);

    proj_gemm<<<dim3(M / 16), dim3(256), 0, stream>>>(accimg, pwb, proj_b, out);
}
